// GRUSingleRollout_80857054314866
// MI455X (gfx1250) — matrix operand order repaired and checked by reading, not compiled, not run on hardware
//
#include <hip/hip_runtime.h>
#include <stdint.h>

// ---------------------------------------------------------------------------
// CDNA5 (gfx1250) wave32 WMMA implementation of the bidirectional 2-layer GRU
// encoder + fuse MLP + head from the reference.
//
// All GEMM-like math runs through V_WMMA_F32_16X16X32_F16 (f16 inputs, f32
// accumulate).  Weights are pre-swizzled once into the per-lane B-fragment
// layout.  Recurrent (Whh) fragments are cached in LDS per block (re-used 50x
// per wave); input-side (Wih) fragments stream from L2 (weights ~1MB << 192MB
// L2, so always resident).
// ---------------------------------------------------------------------------

typedef _Float16 v16h __attribute__((ext_vector_type(16)));
typedef float    v8f  __attribute__((ext_vector_type(8)));

union V16H { v16h v; uint4 q[2]; _Float16 h[16]; };

static __device__ __forceinline__ v8f wmma32f16(v16h a, v16h b, v8f c) {
  // D = A(16x32 f16) * B(32x16 f16) + C(16x16 f32)
  return __builtin_amdgcn_wmma_f32_16x16x32_f16(false, a, false, b, (short)0, c,
                                                false, false);
}

// B fragment: pre-swizzled as frags[fragIdx][lane][16 halves] (global or LDS).
static __device__ __forceinline__ v16h load_bfrag(const _Float16* __restrict__ frags,
                                                  int fragIdx, int lane) {
  const uint4* p = (const uint4*)(frags + ((size_t)fragIdx * 32 + lane) * 16);
  V16H u;
  u.q[0] = p[0];
  u.q[1] = p[1];
  return u.v;
}

// A fragment gather from a row-major f16 row (rowBase = row m for this lane).
// k order used here (contiguous, as in swizzle_kernel): lane<16 holds K = c*32 + [0..15];
//                  lane>=16 holds K = c*32 + [16..31].
static __device__ __forceinline__ v16h load_afrag(const _Float16* __restrict__ rowBase,
                                                  int chunk, int lhi) {
  const int k0 = chunk * 32 + lhi * 16;
  V16H u;
  u.q[0] = *(const uint4*)(rowBase + k0);
  u.q[1] = *(const uint4*)(rowBase + k0 + 8);
  return u.v;
}

static __device__ __forceinline__ float fsigmoid(float x) {
  return 1.f / (1.f + __expf(-x));
}
static __device__ __forceinline__ float ftanh(float x) {
  return 1.f - 2.f / (__expf(2.f * x) + 1.f);  // stable at +/-inf
}
static __device__ __forceinline__ v8f splat8(float x) {
  v8f r;
#pragma unroll
  for (int i = 0; i < 8; ++i) r[i] = x;
  return r;
}

// ---------------------------------------------------------------------------
// Sequence lengths from boolean masks (prefix-true masks; len = popcount).
// JAX bool arrays are 1 byte/element.
// ---------------------------------------------------------------------------
__global__ void lens_kernel(const uint8_t* __restrict__ mask,
                            int* __restrict__ lens, int S, int T) {
  int s = blockIdx.x * blockDim.x + threadIdx.x;
  if (s >= S) return;
  int c = 0;
  for (int t = 0; t < T; ++t) c += mask[(size_t)s * T + t] ? 1 : 0;
  lens[s] = c;
}

// ---------------------------------------------------------------------------
// Pre-swizzle a row-major f32 weight matrix W[(dir,) Nrows, K] into f16 WMMA
// B-fragments: frag(j,c) covers output cols 16j..16j+15, K-chunk c (32 wide).
// Per-lane layout: n = j*16 + (lane&15); k = c*32 + (lane>>4)*16 + [0..15].
// grid = (Ntiles*Kchunks, ndirs), block = 32.
// ---------------------------------------------------------------------------
__global__ __launch_bounds__(32) void swizzle_kernel(
    const float* __restrict__ W, _Float16* __restrict__ out,
    int Nrows, int K, int Kchunks, int fragsPerDir) {
  const int lane = threadIdx.x;
  const int frag = blockIdx.x;
  const int d    = blockIdx.y;
  const float* Wd = W + (size_t)d * (size_t)Nrows * K;
  _Float16* dst = out + ((size_t)(d * fragsPerDir + frag) * 32 + lane) * 16;
  const int j  = frag / Kchunks;
  const int c  = frag % Kchunks;
  const int n  = j * 16 + (lane & 15);
  const int kb = c * 32 + (lane >> 4) * 16;
#pragma unroll
  for (int i = 0; i < 16; ++i) {
    int k = kb + i;
    float v = (n < Nrows && k < K) ? Wd[(size_t)n * K + k] : 0.f;
    dst[i] = (_Float16)v;
  }
}

// ---------------------------------------------------------------------------
// One GRU layer, both directions.  Block = 64 threads = 2 waves; wave == dir.
// Each wave owns a 16-sequence tile: per step it computes the 16x192 gate
// pre-activations with WMMA, applies the GRU cell elementwise in the C-layout
// (lane = output col within 16-tile, vgpr = sample row), masks padded steps,
// and stages h back to LDS as f16 for the next step's A-fragments.
// The 24 Whh B-fragments per direction (24KB) are cached in LDS at start.
//   DC = 1: layer 0 (x f32, Din=2 zero-padded into one K=32 chunk)
//   DC = 4: layer 1 (x f16 from workspace, Din=128)
// ---------------------------------------------------------------------------
template <int DC>
__global__ __launch_bounds__(64) void gru_layer_kernel(
    const float* __restrict__ x32, const _Float16* __restrict__ x16,
    const int* __restrict__ lens, int T,
    const _Float16* __restrict__ wihF0, int wihFragsPerDir,
    const _Float16* __restrict__ whhF0,
    const float* __restrict__ bias,            // (2, 2, 192): [dir][ih/hh][3H]
    _Float16* __restrict__ seq_out,            // (S, T, 128) or null
    _Float16* __restrict__ h_out) {            // (S, 128) or null
  __shared__ __align__(16) _Float16 hstage[2][16][64];     // 4 KB
  __shared__ __align__(16) _Float16 whhL[2][24 * 512];     // 48 KB

  const int lane = threadIdx.x & 31;
  const int dir  = threadIdx.x >> 5;
  const int lrow = lane & 15;
  const int lhi  = lane >> 4;
  const int tile = blockIdx.x;
  const int seqA = tile * 16 + lrow;           // A-matrix row m = lrow

  int lenv[8];                                  // lens for C-layout rows
#pragma unroll
  for (int v = 0; v < 8; ++v) lenv[v] = lens[tile * 16 + v + 8 * lhi];

  const float* bih = bias + dir * 2 * 192;
  const float* bhh = bih + 192;
  const _Float16* wihF = wihF0 + (size_t)dir * wihFragsPerDir * 512;

  // Cache this direction's Whh fragments in LDS (wave-private region: no sync).
  {
    const uint4* src = (const uint4*)(whhF0 + (size_t)dir * 24 * 512);
    uint4* dst = (uint4*)&whhL[dir][0];
#pragma unroll 4
    for (int i = 0; i < 48; ++i) dst[i * 32 + lane] = src[i * 32 + lane];
  }
  const _Float16* whhF = &whhL[dir][0];

  float h[4][8];
#pragma unroll
  for (int j = 0; j < 4; ++j)
#pragma unroll
    for (int v = 0; v < 8; ++v) {
      h[j][v] = 0.f;
      hstage[dir][v + 8 * lhi][16 * j + lrow] = (_Float16)0.f;
    }

  float br[4], bz[4], bgn[4], bhn[4];
#pragma unroll
  for (int j = 0; j < 4; ++j) {
    int n  = 16 * j + lrow;
    br[j]  = bih[n] + bhh[n];                  // r gate: both biases fold
    bz[j]  = bih[64 + n] + bhh[64 + n];        // z gate: both biases fold
    bgn[j] = bih[128 + n];                     // n gate x-side
    bhn[j] = bhh[128 + n];                     // n gate h-side (scaled by r)
  }

  for (int step = 0; step < T; ++step) {
    const int t = dir ? (T - 1 - step) : step;

    // ---- x A-fragments -------------------------------------------------
    v16h xf[DC];
    if constexpr (DC == 4) {
      const _Float16* xrow = x16 + ((size_t)seqA * T + t) * 128;
#pragma unroll
      for (int c = 0; c < 4; ++c) xf[c] = load_afrag(xrow, c, lhi);
      if (step + 1 < T) {
        int t2 = dir ? (T - 2 - step) : (step + 1);
        __builtin_prefetch(x16 + ((size_t)seqA * T + t2) * 128, 0, 1);
      }
    } else {
      V16H u;
#pragma unroll
      for (int i = 0; i < 16; ++i) u.h[i] = (_Float16)0.f;
      if (lhi == 0) {                          // K=0,1 live in lanes 0..15
        const float* xp = x32 + ((size_t)seqA * T + t) * 2;
        u.h[0] = (_Float16)xp[0];
        u.h[1] = (_Float16)xp[1];
      }
      xf[0] = u.v;
    }

    // ---- h A-fragments (prev state, f16, from LDS) ---------------------
    v16h hf[2];
#pragma unroll
    for (int c = 0; c < 2; ++c) hf[c] = load_afrag(&hstage[dir][lrow][0], c, lhi);

    // ---- gates per 16-wide hidden chunk j ------------------------------
#pragma unroll
    for (int j = 0; j < 4; ++j) {
      v8f ar = splat8(br[j]), az = splat8(bz[j]);
      v8f agn = splat8(bgn[j]), ahn = splat8(bhn[j]);
#pragma unroll
      for (int c = 0; c < DC; ++c) {
        ar  = wmma32f16(xf[c], load_bfrag(wihF, (0 + j) * DC + c, lane), ar);
        az  = wmma32f16(xf[c], load_bfrag(wihF, (4 + j) * DC + c, lane), az);
        agn = wmma32f16(xf[c], load_bfrag(wihF, (8 + j) * DC + c, lane), agn);
      }
#pragma unroll
      for (int c = 0; c < 2; ++c) {
        ar  = wmma32f16(hf[c], load_bfrag(whhF, (0 + j) * 2 + c, lane), ar);
        az  = wmma32f16(hf[c], load_bfrag(whhF, (4 + j) * 2 + c, lane), az);
        ahn = wmma32f16(hf[c], load_bfrag(whhF, (8 + j) * 2 + c, lane), ahn);
      }
#pragma unroll
      for (int v = 0; v < 8; ++v) {
        float r  = fsigmoid(ar[v]);
        float z  = fsigmoid(az[v]);
        float nn = ftanh(agn[v] + r * ahn[v]);
        float hp = h[j][v];
        float hn = (1.f - z) * nn + z * hp;
        hp = (t < lenv[v]) ? hn : hp;          // freeze on padded steps
        h[j][v] = hp;
        hstage[dir][v + 8 * lhi][16 * j + lrow] = (_Float16)hp;
        if (seq_out) {
          size_t s = (size_t)(tile * 16 + v + 8 * lhi);
          seq_out[(s * T + t) * 128 + dir * 64 + 16 * j + lrow] = (_Float16)hp;
        }
      }
    }
  }

  if (h_out) {
#pragma unroll
    for (int j = 0; j < 4; ++j)
#pragma unroll
      for (int v = 0; v < 8; ++v)
        h_out[(size_t)(tile * 16 + v + 8 * lhi) * 128 + dir * 64 + 16 * j + lrow] =
            (_Float16)h[j][v];
  }
}

// ---------------------------------------------------------------------------
// Fuse MLP (256->128 relu, 128->128 relu) + head (128->120), all WMMA.
// One wave per 16-sample tile; activations staged through LDS to convert
// C-layout -> A-layout between layers.
// ---------------------------------------------------------------------------
__global__ __launch_bounds__(32) void fuse_head_kernel(
    const _Float16* __restrict__ hag, const _Float16* __restrict__ hego,
    const _Float16* __restrict__ w1f, const float* __restrict__ b1,
    const _Float16* __restrict__ w2f, const float* __restrict__ b2,
    const _Float16* __restrict__ whf, const float* __restrict__ bh,
    float* __restrict__ out, int Na) {
  __shared__ __align__(16) _Float16 zstage[16][128];

  const int lane = threadIdx.x & 31;
  const int lrow = lane & 15;
  const int lhi  = lane >> 4;
  const int tile = blockIdx.x;
  const int seqA = tile * 16 + lrow;

  // z = [h_ag(128) ; h_ego_broadcast(128)] -> 8 A chunks (K=256)
  v16h a0[8];
  const _Float16* hr = hag + (size_t)seqA * 128;
#pragma unroll
  for (int c = 0; c < 4; ++c) a0[c] = load_afrag(hr, c, lhi);
  const _Float16* er = hego + (size_t)(seqA / Na) * 128;
#pragma unroll
  for (int c = 0; c < 4; ++c) a0[4 + c] = load_afrag(er, c, lhi);

  for (int j = 0; j < 8; ++j) {
    v8f acc = splat8(b1[16 * j + lrow]);
#pragma unroll
    for (int c = 0; c < 8; ++c)
      acc = wmma32f16(a0[c], load_bfrag(w1f, j * 8 + c, lane), acc);
#pragma unroll
    for (int v = 0; v < 8; ++v)
      zstage[v + 8 * lhi][16 * j + lrow] = (_Float16)fmaxf(acc[v], 0.f);
  }

  v16h a1[4];
#pragma unroll
  for (int c = 0; c < 4; ++c) a1[c] = load_afrag(&zstage[lrow][0], c, lhi);

  for (int j = 0; j < 8; ++j) {
    v8f acc = splat8(b2[16 * j + lrow]);
#pragma unroll
    for (int c = 0; c < 4; ++c)
      acc = wmma32f16(a1[c], load_bfrag(w2f, j * 4 + c, lane), acc);
#pragma unroll
    for (int v = 0; v < 8; ++v)
      zstage[v + 8 * lhi][16 * j + lrow] = (_Float16)fmaxf(acc[v], 0.f);
  }

  v16h a2[4];
#pragma unroll
  for (int c = 0; c < 4; ++c) a2[c] = load_afrag(&zstage[lrow][0], c, lhi);

  for (int j = 0; j < 8; ++j) {
    const int n = 16 * j + lrow;               // head output col (120 wide)
    v8f acc = splat8(n < 120 ? bh[n] : 0.f);
#pragma unroll
    for (int c = 0; c < 4; ++c)
      acc = wmma32f16(a2[c], load_bfrag(whf, j * 4 + c, lane), acc);
    if (n < 120) {
#pragma unroll
      for (int v = 0; v < 8; ++v)
        out[(size_t)(tile * 16 + v + 8 * lhi) * 120 + n] = acc[v];
    }
  }
}

// ---------------------------------------------------------------------------
// Host launcher
// ---------------------------------------------------------------------------
extern "C" void kernel_launch(void* const* d_in, const int* in_sizes, int n_in,
                              void* d_out, int out_size, void* d_ws, size_t ws_size,
                              hipStream_t stream) {
  (void)in_sizes; (void)n_in; (void)out_size; (void)ws_size;
  const int B = 64, Na = 128, Th = 50;
  const int Sag = B * Na;  // 8192 agent sequences

  char* wp = (char*)d_ws;
  auto take = [&](size_t bytes) -> char* {
    char* p = wp;
    wp += (bytes + 255) & ~(size_t)255;
    return p;
  };

  _Float16* x1_ag  = (_Float16*)take((size_t)Sag * Th * 128 * 2);  // ~100 MB
  _Float16* x1_ego = (_Float16*)take((size_t)B * Th * 128 * 2);
  _Float16* hag    = (_Float16*)take((size_t)Sag * 128 * 2);
  _Float16* hego   = (_Float16*)take((size_t)B * 128 * 2);
  int* lens_ag     = (int*)take((size_t)Sag * 4);
  int* lens_ego    = (int*)take((size_t)B * 4);
  _Float16* fA0ih  = (_Float16*)take(2 * 12 * 1024);
  _Float16* fA0hh  = (_Float16*)take(2 * 24 * 1024);
  _Float16* fA1ih  = (_Float16*)take(2 * 48 * 1024);
  _Float16* fA1hh  = (_Float16*)take(2 * 24 * 1024);
  _Float16* fE0ih  = (_Float16*)take(2 * 12 * 1024);
  _Float16* fE0hh  = (_Float16*)take(2 * 24 * 1024);
  _Float16* fE1ih  = (_Float16*)take(2 * 48 * 1024);
  _Float16* fE1hh  = (_Float16*)take(2 * 24 * 1024);
  _Float16* fW1    = (_Float16*)take(64 * 1024);
  _Float16* fW2    = (_Float16*)take(32 * 1024);
  _Float16* fWH    = (_Float16*)take(32 * 1024);

  // Sequence lengths
  lens_kernel<<<(Sag + 255) / 256, 256, 0, stream>>>(
      (const uint8_t*)d_in[1], lens_ag, Sag, Th);
  lens_kernel<<<1, 64, 0, stream>>>((const uint8_t*)d_in[3], lens_ego, B, Th);

  // Weight fragment swizzles (W[dir, 3H, K] -> per-lane B fragments)
  swizzle_kernel<<<dim3(12, 2), 32, 0, stream>>>((const float*)d_in[4],  fA0ih, 192, 2,   1, 12);
  swizzle_kernel<<<dim3(24, 2), 32, 0, stream>>>((const float*)d_in[5],  fA0hh, 192, 64,  2, 24);
  swizzle_kernel<<<dim3(48, 2), 32, 0, stream>>>((const float*)d_in[7],  fA1ih, 192, 128, 4, 48);
  swizzle_kernel<<<dim3(24, 2), 32, 0, stream>>>((const float*)d_in[8],  fA1hh, 192, 64,  2, 24);
  swizzle_kernel<<<dim3(12, 2), 32, 0, stream>>>((const float*)d_in[10], fE0ih, 192, 2,   1, 12);
  swizzle_kernel<<<dim3(24, 2), 32, 0, stream>>>((const float*)d_in[11], fE0hh, 192, 64,  2, 24);
  swizzle_kernel<<<dim3(48, 2), 32, 0, stream>>>((const float*)d_in[13], fE1ih, 192, 128, 4, 48);
  swizzle_kernel<<<dim3(24, 2), 32, 0, stream>>>((const float*)d_in[14], fE1hh, 192, 64,  2, 24);
  swizzle_kernel<<<dim3(64, 1), 32, 0, stream>>>((const float*)d_in[16], fW1, 128, 256, 8, 64);
  swizzle_kernel<<<dim3(32, 1), 32, 0, stream>>>((const float*)d_in[18], fW2, 128, 128, 4, 32);
  swizzle_kernel<<<dim3(32, 1), 32, 0, stream>>>((const float*)d_in[20], fWH, 120, 128, 4, 32);

  // Agents: layer 0 -> x1_ag, layer 1 -> hag
  gru_layer_kernel<1><<<Sag / 16, 64, 0, stream>>>(
      (const float*)d_in[0], nullptr, lens_ag, Th,
      fA0ih, 12, fA0hh, (const float*)d_in[6], x1_ag, nullptr);
  gru_layer_kernel<4><<<Sag / 16, 64, 0, stream>>>(
      nullptr, x1_ag, lens_ag, Th,
      fA1ih, 48, fA1hh, (const float*)d_in[9], nullptr, hag);

  // Ego: layer 0 -> x1_ego, layer 1 -> hego
  gru_layer_kernel<1><<<B / 16, 64, 0, stream>>>(
      (const float*)d_in[2], nullptr, lens_ego, Th,
      fE0ih, 12, fE0hh, (const float*)d_in[12], x1_ego, nullptr);
  gru_layer_kernel<4><<<B / 16, 64, 0, stream>>>(
      nullptr, x1_ego, lens_ego, Th,
      fE1ih, 48, fE1hh, (const float*)d_in[15], nullptr, hego);

  // Fuse MLP + head -> (B*Na, 120) f32
  fuse_head_kernel<<<Sag / 16, 32, 0, stream>>>(
      hag, hego, fW1, (const float*)d_in[17], fW2, (const float*)d_in[19],
      fWH, (const float*)d_in[21], (float*)d_out, Na);
}
